// ScaledDotProductAttention_69226282876939
// MI455X (gfx1250) — matrix operand order repaired and checked by reading, not compiled, not run on hardware
//
#include <hip/hip_runtime.h>
#include <cstdint>
#include <cstddef>

// CDNA5 (gfx1250) fused scaled-dot-product attention, wave32 + WMMA bf16,
// double-buffered async global->LDS staging (ASYNCcnt pipeline).
//   grid = (L/16, B), block = 256 threads (8 waves)
//   d_out = [B,L,D] output  ++  [B,L,L] attention probabilities

typedef __attribute__((ext_vector_type(16))) __bf16 v16bf;
typedef __attribute__((ext_vector_type(8)))  float  v8f;
typedef int v4i_vs __attribute__((vector_size(16)));   // matches builtin param type

#define B_    64
#define L_    2048
#define D_    64
#define SROW  2064   // padded row stride (floats) for the LDS score buffer
// LDS: scores 16*SROW*4 | Q bf16 16*64*2 | per-wave double slab 8*2048*4 | O-acc | inv
#define SMEM_BYTES (16*SROW*4 + 16*64*2 + 8*2048*4 + 4*16*16*4 + 16*4)

// ---- async global->LDS staging (ASYNCcnt path), guarded for toolchain drift ----
#if __has_builtin(__builtin_amdgcn_global_load_async_to_lds_b128)
#define HAVE_ASYNC_LDS 1
__device__ __forceinline__ void async_copy_b128(const void* g, void* l) {
  __builtin_amdgcn_global_load_async_to_lds_b128(
      (__attribute__((address_space(1))) v4i_vs*)(uintptr_t)g,
      (__attribute__((address_space(3))) v4i_vs*)(uint32_t)(uintptr_t)l,
      0, 0);
}
#define ASYNC_WAIT_LE0() asm volatile("s_wait_asynccnt 0" ::: "memory")
#define ASYNC_WAIT_LE4() asm volatile("s_wait_asynccnt 4" ::: "memory")
#define ASYNC_WAIT_LE8() asm volatile("s_wait_asynccnt 8" ::: "memory")
#else
#define HAVE_ASYNC_LDS 0
#define ASYNC_WAIT_LE0()
#define ASYNC_WAIT_LE4()
#define ASYNC_WAIT_LE8()
#endif

// 16-bit A-matrix 16x32 fragment K-pattern (ISA 7.12.2):
// lanes 0-15: h0..7 -> K=0..7, h8..15 -> K=16..23 ; lanes 16-31: +8
__device__ __forceinline__ int a_kpat(int h, int lane) {
  int base = (h < 8) ? ((lane < 16) ? 0 : 8) : ((lane < 16) ? 16 : 24);
  return base + (h & 7);
}
// 16-bit B-matrix 32x16 fragment K-pattern (must match A):
// lanes 0-15: h0..7 -> K=0..7, h8..15 -> K=16..23 ; lanes 16-31: +8
__device__ __forceinline__ int b_kpat(int h, int lane) {
  return ((h < 8) ? 0 : 16) + ((lane < 16) ? 0 : 8) + (h & 7);
}

__global__ __launch_bounds__(256)
void sdpa_wmma_kernel(const float* __restrict__ Q, const float* __restrict__ K,
                      const float* __restrict__ V, const uint8_t* __restrict__ M,
                      float* __restrict__ Out, float* __restrict__ Attn) {
  extern __shared__ unsigned char smem[];
  float*  sS   = (float*)smem;                               // 16 x SROW
  __bf16* sQ   = (__bf16*)(smem + 16 * SROW * 4);            // 16 x 64
  float*  sStg = (float*)(smem + 16 * SROW * 4 + 2048);      // 8 x 2048 (2 slabs/wave)
  float*  sAcc = sStg + 8 * 2048;                            // 4 x 16 x 16
  float*  sInv = sAcc + 4 * 16 * 16;                         // 16

  const int tid  = threadIdx.x;
  const int wv   = tid >> 5;
  const int lane = tid & 31;
  const int qt   = blockIdx.x;      // 16-row query tile
  const int b    = blockIdx.y;      // batch

  const float*   Qb = Q + ((size_t)b * L_ + qt * 16) * D_;
  const float*   Kb = K + (size_t)b * L_ * D_;
  const float*   Vb = V + (size_t)b * L_ * D_;
  const uint8_t* Mb = M + ((size_t)b * L_ + qt * 16) * L_;
  float* Ob = Out  + ((size_t)b * L_ + qt * 16) * D_;
  float* Ab = Attn + ((size_t)b * L_ + qt * 16) * L_;

  float* slab0 = sStg + wv * 2048;               // this wave's double buffer
  float* slab1 = slab0 + 1024;
  const int khalf = wv >> 2, n4 = wv & 3;        // phase-4 work split

  // ---------- Phase 0: Q tile -> LDS (bf16) ----------
  for (int i = tid; i < 16 * 64; i += 256) sQ[i] = (__bf16)Qb[i];
  __syncthreads();

  // Persistent Q A-fragments (K = 0..31 and 32..63), identical in every wave.
  v16bf aQ0, aQ1;
  #pragma unroll
  for (int h = 0; h < 16; ++h) {
    int m = lane & 15;
    int p = a_kpat(h, lane);
    aQ0[h] = sQ[m * 64 + p];
    aQ1[h] = sQ[m * 64 + 32 + p];
  }

  // ---------- Phase 1: S = QK^T / 8, masked -> LDS score buffer ----------
#if HAVE_ASYNC_LDS
  // preload first K tile (16x64 f32 = 4 KB), 16 B per lane per op
  #pragma unroll
  for (int i = 0; i < 8; ++i) {
    int c = i * 32 + lane;
    async_copy_b128((const char*)Kb + (size_t)wv * 4096 + (size_t)c * 16,
                    (char*)slab0 + c * 16);
  }
#endif
  for (int iter = 0; iter < 16; ++iter) {        // 16 key tiles per wave, uniform
    int kt = wv + iter * 8;
    float* cur = (iter & 1) ? slab1 : slab0;
#if HAVE_ASYNC_LDS
    if (iter + 1 < 16) {                         // pipeline: issue next, then wait cur
      float* nxt = (iter & 1) ? slab0 : slab1;
      #pragma unroll
      for (int i = 0; i < 8; ++i) {
        int c = i * 32 + lane;
        async_copy_b128((const char*)Kb + (size_t)(kt + 8) * 4096 + (size_t)c * 16,
                        (char*)nxt + c * 16);
      }
      ASYNC_WAIT_LE8();                          // in-order: cur's 8 copies done
    } else {
      ASYNC_WAIT_LE0();
    }
#else
    #pragma unroll
    for (int i = 0; i < 32; ++i) {
      int idx = lane + i * 32;                   // 0..1023 == row*64+col
      cur[idx] = Kb[(size_t)(kt * 16) * D_ + idx];
    }
#endif
    v16bf bK0, bK1;                              // B = K^T (32x16), column n = key row
    #pragma unroll
    for (int h = 0; h < 16; ++h) {
      int n = lane & 15;
      int p = b_kpat(h, lane);
      bK0[h] = (__bf16)cur[n * 64 + p];
      bK1[h] = (__bf16)cur[n * 64 + 32 + p];
    }
    v8f c = {};
    c = __builtin_amdgcn_wmma_f32_16x16x32_bf16(false, aQ0, false, bK0, (short)0, c, false, false);
    c = __builtin_amdgcn_wmma_f32_16x16x32_bf16(false, aQ1, false, bK1, (short)0, c, false, false);

    #pragma unroll
    for (int i = 0; i < 8; ++i) {                // C layout: m = i (+8 hi-half), n = lane&15
      int m = i + ((lane < 16) ? 0 : 8);
      int n = lane & 15;
      int col = kt * 16 + n;
      float s = c[i] * 0.125f;                   // 1 / TEMPERATURE
      // mask is a read-once 268 MB stream: keep it out of L2 (NT load)
      if (__builtin_nontemporal_load(Mb + (size_t)m * L_ + col)) s = -1e10f;
      sS[m * SROW + col] = s;
    }
  }

#if HAVE_ASYNC_LDS
  // prefetch first V block (32x16 f32 = 2 KB) so softmax hides its latency
  #pragma unroll
  for (int i = 0; i < 4; ++i) {
    int c = i * 32 + lane;                       // 16-byte chunk index, 0..127
    int row = c >> 2, grp = c & 3;
    async_copy_b128((const char*)Vb +
                        ((size_t)(khalf * 1024 + row) * D_ + n4 * 16 + grp * 4) * 4,
                    (char*)slab0 + c * 16);
  }
#endif
  __syncthreads();

  // ---------- Phase 2: row softmax statistics (16 threads per row) ----------
  {
    int r = tid >> 4, sub = tid & 15;            // row r owned by one half-wave
    float mx = -3.0e38f;
    for (int c = sub; c < L_; c += 16) mx = fmaxf(mx, sS[r * SROW + c]);
    #pragma unroll
    for (int d = 8; d >= 1; d >>= 1) mx = fmaxf(mx, __shfl_xor(mx, d, 32));
    float sum = 0.f;
    for (int c = sub; c < L_; c += 16) {
      float e = __expf(sS[r * SROW + c] - mx);
      sS[r * SROW + c] = e;
      sum += e;
    }
    #pragma unroll
    for (int d = 8; d >= 1; d >>= 1) sum += __shfl_xor(sum, d, 32);
    if (sub == 0) sInv[r] = 1.0f / sum;
  }
  __syncthreads();

  // ---------- Phase 3: normalize P in LDS + stream attn to HBM (NT stores) ----------
  for (int idx = tid; idx < 16 * L_; idx += 256) {
    int m = idx >> 11, c = idx & (L_ - 1);
    float p = sS[m * SROW + c] * sInv[m];
    sS[m * SROW + c] = p;
    // attn is a write-once 1.07 GB stream: NT store keeps L2 hot for K/V reuse
    __builtin_nontemporal_store(p, Ab + (size_t)m * L_ + c);
  }
  __syncthreads();

  // ---------- Phase 4: O = P * V  (waves: 4 N-tiles x 2 K-halves) ----------
  v8f o = {};
  for (int s = 0; s < 32; ++s) {                 // uniform trip count, EXEC all-ones
    int kb = khalf * 1024 + s * 32;
    float* cur = (s & 1) ? slab1 : slab0;
#if HAVE_ASYNC_LDS
    if (s + 1 < 32) {                            // pipeline next V block
      float* nxt = (s & 1) ? slab0 : slab1;
      #pragma unroll
      for (int i = 0; i < 4; ++i) {
        int c = i * 32 + lane;
        int row = c >> 2, grp = c & 3;
        async_copy_b128((const char*)Vb +
                            ((size_t)(kb + 32 + row) * D_ + n4 * 16 + grp * 4) * 4,
                        (char*)nxt + c * 16);
      }
      ASYNC_WAIT_LE4();                          // in-order: cur's 4 copies done
    } else {
      ASYNC_WAIT_LE0();
    }
#else
    #pragma unroll
    for (int i = 0; i < 16; ++i) {
      int idx = lane + i * 32;                   // 0..511
      int row = idx >> 4, col = idx & 15;
      cur[idx] = Vb[(size_t)(kb + row) * D_ + n4 * 16 + col];
    }
#endif
    v16bf aP, bV;
    #pragma unroll
    for (int h = 0; h < 16; ++h) {
      int pa = a_kpat(h, lane);
      int pb = b_kpat(h, lane);
      aP[h] = (__bf16)sS[(lane & 15) * SROW + kb + pa];
      bV[h] = (__bf16)cur[pb * 16 + (lane & 15)];
    }
    o = __builtin_amdgcn_wmma_f32_16x16x32_bf16(false, aP, false, bV, (short)0, o, false, false);
  }
  // cross-wave K-half reduction through LDS, then O store
  if (wv >= 4) {
    #pragma unroll
    for (int i = 0; i < 8; ++i) {
      int m = i + ((lane < 16) ? 0 : 8);
      sAcc[(n4 * 16 + m) * 16 + (lane & 15)] = o[i];
    }
  }
  __syncthreads();
  if (wv < 4) {
    #pragma unroll
    for (int i = 0; i < 8; ++i) {
      int m = i + ((lane < 16) ? 0 : 8);
      int n = lane & 15;
      float val = o[i] + sAcc[(n4 * 16 + m) * 16 + n];
      Ob[(size_t)m * D_ + n4 * 16 + n] = val;
    }
  }
}

extern "C" void kernel_launch(void* const* d_in, const int* in_sizes, int n_in,
                              void* d_out, int out_size, void* d_ws, size_t ws_size,
                              hipStream_t stream) {
  (void)in_sizes; (void)n_in; (void)out_size; (void)d_ws; (void)ws_size;
  const float*   q    = (const float*)d_in[0];
  const float*   k    = (const float*)d_in[1];
  const float*   v    = (const float*)d_in[2];
  const uint8_t* mask = (const uint8_t*)d_in[3];   // jax bool -> 1 byte/elem
  float* out  = (float*)d_out;
  float* attn = out + (size_t)B_ * L_ * D_;        // tuple: (output, attn) flat

  dim3 grid(L_ / 16, B_), block(256);
  sdpa_wmma_kernel<<<grid, block, SMEM_BYTES, stream>>>(q, k, v, mask, out, attn);
}
